// SAN_decoder_47571057770747
// MI455X (gfx1250) — matrix operand order repaired and checked by reading, not compiled, not run on hardware
//
#include <hip/hip_runtime.h>
#include <math.h>

// ---------------------------------------------------------------------------
// SAN decoder forward for MI455X (gfx1250). All GEMMs use bf16 WMMA
// v_wmma_f32_16x16x32_bf16 with fp32 accumulation; nonlinearities / softmax /
// alpha-sum state stay fp32. The 11x11 coverage conv + K->A projection are
// fused into a single (121 x 256) weight so the per-step attention becomes
// one M=8192,K=128,N=256 WMMA GEMM. Each wave computes a 16x32 output tile
// (two accumulators sharing one A fragment) for 2x WMMA density per loop.
// ---------------------------------------------------------------------------

typedef __bf16 bf16;
typedef __attribute__((ext_vector_type(8)))  bf16  v8bf;
typedef __attribute__((ext_vector_type(16))) bf16  v16bf;
typedef __attribute__((ext_vector_type(8)))  float v8f;

constexpr int NB   = 16;    // batch
constexpr int NT   = 100;   // timesteps
constexpr int NC   = 684;   // cnn channels
constexpr int NCP  = 704;   // padded to multiple of 32
constexpr int NH   = 256;   // hidden
constexpr int NH3  = 768;   // 3*hidden (GRU gates)
constexpr int NA   = 256;   // attention dim
constexpr int NV   = 115;   // vocab
constexpr int NS   = 7;     // struct classes
constexpr int NHW  = 512;   // 16 x 32 feature positions
constexpr int NTP  = 128;   // 121 conv taps padded to 128

// ------------------------------ utility kernels ----------------------------

__global__ void k_zero(float* p, int n) {
  for (int i = blockIdx.x * blockDim.x + threadIdx.x; i < n;
       i += gridDim.x * blockDim.x) p[i] = 0.f;
}

__global__ void k_cvt(const float* __restrict__ s, bf16* __restrict__ d, int n) {
  int i = blockIdx.x * blockDim.x + threadIdx.x;
  if (i < n) d[i] = (bf16)s[i];
}

// transpose (N,K) fp32 weight -> (Kp,Np) bf16, zero padded
__global__ void k_prep_wt(const float* __restrict__ W, bf16* __restrict__ Wt,
                          int N, int K, int Kp, int Np) {
  int i = blockIdx.x * blockDim.x + threadIdx.x;
  if (i >= Kp * Np) return;
  int k = i / Np, n = i - k * Np;
  Wt[i] = (k < K && n < N) ? (bf16)W[n * K + k] : (bf16)0.f;
}

// fusedW[t,a] = sum_k cov_k[k,t] * cov_w[a,k]   (t in 0..120, padded to 128)
__global__ void k_fuse_cov(const float* __restrict__ ck,
                           const float* __restrict__ cw,
                           bf16* __restrict__ out) {
  int i = blockIdx.x * blockDim.x + threadIdx.x;
  if (i >= NTP * NA) return;
  int t = i / NA, a = i - t * NA;
  float s = 0.f;
  if (t < 121) {
    for (int k = 0; k < 256; ++k) s += ck[k * 121 + t] * cw[a * 256 + k];
  }
  out[i] = (bf16)s;
}

// cnn (B,C,HW) fp32 -> Xc (B*HW, NCP) bf16 row-major, zero padded
__global__ void k_xc(const float* __restrict__ cnn, bf16* __restrict__ Xc) {
  int i = blockIdx.x * blockDim.x + threadIdx.x;
  if (i >= NB * NHW * NCP) return;
  int pos = i / NCP, c = i - pos * NCP;
  int b = pos >> 9, hw = pos & 511;
  Xc[i] = (c < NC) ? (bf16)cnn[(b * NC + c) * NHW + hw] : (bf16)0.f;
}

// strided mask sample + per-batch mask sum
__global__ void k_mask(const float* __restrict__ im, float* __restrict__ maskS,
                       float* __restrict__ msum) {
  __shared__ float red[512];
  int b = blockIdx.x, hw = threadIdx.x, y = hw >> 5, x = hw & 31;
  float v = im[(b * 256 + y * 16) * 512 + x * 16];
  maskS[b * NHW + hw] = v;
  red[hw] = v; __syncthreads();
  for (int s = 256; s > 0; s >>= 1) {
    if (hw < s) red[hw] += red[hw + s];
    __syncthreads();
  }
  if (!hw) msum[b] = red[0];
}

// avg[b,c] = sum_hw cnn*maskS / msum[b]  -> bf16 (B, NCP) padded
__global__ void k_avg(const float* __restrict__ cnn, const float* __restrict__ maskS,
                      const float* __restrict__ msum, bf16* __restrict__ avg_bf) {
  int i = blockIdx.x * blockDim.x + threadIdx.x;
  if (i >= NB * NCP) return;
  int b = i / NCP, c = i - b * NCP;
  float s = 0.f;
  if (c < NC) {
    const float* fp = cnn + (b * NC + c) * NHW;
    const float* mp = maskS + b * NHW;
    for (int hw = 0; hw < NHW; ++hw) s += fp[hw] * mp[hw];
    s /= msum[b];
  }
  avg_bf[i] = (bf16)s;
}

// ------------------------------ WMMA GEMM ----------------------------------
// Y(M x N, ld=ldy, fp32) = X(M x Kp, bf16) @ Wt(Kp x Npad=ldw, bf16) [+bias]
// One wave per 16x32 tile (two accumulators share one A fragment).
// Fragment layouts per CDNA5 ISA 7.12.2:
//   A 16x32 bf16: lane<16 -> row=lane, K {0..7,16..23}; lane>=16 -> K {8..15,24..31}
//   B 32x16 bf16: lane -> col n=lane%16, K same as A (lane<16 {0..7,16..23}; lane>=16 {8..15,24..31})
//   C/D f32: vgpr v, lane<16 -> (m=v, n=lane); lane>=16 -> (m=v+8, n=lane-16)
// ldw must be a multiple of 32. mode: 0 = store, 1 = accumulate, 2 = tanh+store
__global__ __launch_bounds__(256) void k_gemm(const bf16* __restrict__ X, int ldx,
                                              const bf16* __restrict__ Wt, int ldw,
                                              const float* __restrict__ bias,
                                              float* __restrict__ Y, int ldy,
                                              int M, int N, int Kp, int mode) {
  int lane = threadIdx.x & 31;
  int tile = blockIdx.x * 8 + (threadIdx.x >> 5);
  int ntN = ldw >> 5;                      // 32-wide N tiles
  if (tile >= (M >> 4) * ntN) return;
  int mt = tile / ntN, nt = tile - mt * ntN;
  int mrow  = (mt << 4) + (lane & 15);
  int khalf = (lane < 16) ? 0 : 8;
  int n0    = nt << 5;
  v8f acc0 = {};
  v8f acc1 = {};
  const bf16* xrow = X + mrow * ldx + khalf;
  const bf16* wrow = Wt + khalf * ldw + n0 + (lane & 15);
  for (int kk = 0; kk < Kp; kk += 32) {
    v8bf alo = *(const v8bf*)(xrow + kk);
    v8bf ahi = *(const v8bf*)(xrow + kk + 16);
    v16bf af = __builtin_shufflevector(alo, ahi, 0, 1, 2, 3, 4, 5, 6, 7,
                                       8, 9, 10, 11, 12, 13, 14, 15);
    const bf16* wk = wrow + kk * ldw;
    v16bf bm0, bm1;
#pragma unroll
    for (int s = 0; s < 16; ++s) {
      int ks = (s < 8) ? s : s + 8;
      bm0[s] = wk[ks * ldw];
      bm1[s] = wk[ks * ldw + 16];
    }
    __builtin_prefetch(wk + 32 * ldw, 0, 3);   // speculative, branch-free
    acc0 = __builtin_amdgcn_wmma_f32_16x16x32_bf16(false, af, false, bm0,
                                                   (short)0, acc0, false, false);
    acc1 = __builtin_amdgcn_wmma_f32_16x16x32_bf16(false, af, false, bm1,
                                                   (short)0, acc1, false, false);
  }
  int mbase = (mt << 4) + ((lane < 16) ? 0 : 8);
  int ncol0 = n0 + (lane & 15);
  int ncol1 = ncol0 + 16;
  if (ncol0 < N) {
    float bv = bias ? bias[ncol0] : 0.f;
#pragma unroll
    for (int v = 0; v < 8; ++v) {
      float r = acc0[v] + bv;
      if (mode == 2) r = tanhf(r);
      float* yp = Y + (mbase + v) * ldy + ncol0;
      if (mode == 1) *yp += r; else *yp = r;
    }
  }
  if (ncol1 < N) {
    float bv = bias ? bias[ncol1] : 0.f;
#pragma unroll
    for (int v = 0; v < 8; ++v) {
      float r = acc1[v] + bv;
      if (mode == 2) r = tanhf(r);
      float* yp = Y + (mbase + v) * ldy + ncol1;
      if (mode == 1) *yp += r; else *yp = r;
    }
  }
}

// ------------------------------ decoder kernels ----------------------------

// word-branch gathers: parent hidden, parent alpha-sum, token embedding
__global__ void k_gather_w(const int* __restrict__ labels, const float* __restrict__ ph,
                           const float* __restrict__ was, const float* __restrict__ E,
                           float* __restrict__ phid, bf16* __restrict__ phid_bf,
                           bf16* __restrict__ emb_bf, float* __restrict__ asum_cur,
                           int i) {
  int b = blockIdx.x, t = threadIdx.x;
  int pid = labels[(b * NT + i) * 4 + 2];
  int tok = labels[(b * NT + i) * 4 + 3];
  if (t < NH) {
    float hv = ph[(pid * NB + b) * NH + t];
    phid[b * NH + t]    = hv;
    phid_bf[b * NH + t] = (bf16)hv;
    emb_bf[b * NH + t]  = (bf16)E[tok * NH + t];
  }
  asum_cur[b * NHW + t] = was[(pid * NB + b) * NHW + t];
}

// c2p gathers: child/relation embeddings + old c2p hidden -> bf16
__global__ void k_gather_c2p(const int* __restrict__ labels, const float* __restrict__ E,
                             const float* __restrict__ c2ph, bf16* __restrict__ cat_bf,
                             bf16* __restrict__ ce_bf, bf16* __restrict__ re_bf,
                             bf16* __restrict__ c2ph_bf, int i) {
  int b = blockIdx.x, t = threadIdx.x;
  int j = NT - 1 - i;
  int child = labels[(b * NT + j) * 4 + 1];
  int rel   = labels[(b * NT + j) * 4 + 3];
  bool ind  = ((rel >= 108) && (rel <= 114)) || (rel == 0);
  rel = (child == 2) ? 2 : (ind ? rel : 114);
  float ce = E[child * NH + t], re = E[rel * NH + t];
  ce_bf[b * NH + t] = (bf16)ce;
  re_bf[b * NH + t] = (bf16)re;
  cat_bf[b * 2 * NH + t]      = (bf16)ce;
  cat_bf[b * 2 * NH + NH + t] = (bf16)re;
  c2ph_bf[b * NH + t] = (bf16)c2ph[b * NH + t];
}

// fused GRU gate combine: h = (1-z)*n + z*hprev
__global__ void k_gru(const float* __restrict__ gi, const float* __restrict__ gh,
                      const float* __restrict__ hprev, float* __restrict__ hout,
                      bf16* __restrict__ hbf) {
  int idx = blockIdx.x * blockDim.x + threadIdx.x;
  if (idx >= NB * NH) return;
  int b = idx >> 8, j = idx & 255;
  const float* gib = gi + b * NH3;
  const float* ghb = gh + b * NH3;
  float r = 1.f / (1.f + expf(-(gib[j] + ghb[j])));
  float z = 1.f / (1.f + expf(-(gib[NH + j] + ghb[NH + j])));
  float n = tanhf(gib[2 * NH + j] + r * ghb[2 * NH + j]);
  float h = (1.f - z) * n + z * hprev[idx];
  hout[idx] = h;
  hbf[idx]  = (bf16)h;
}

// im2col of alpha-sum for the fused coverage GEMM: (B*HW, 128) bf16
__global__ void k_im2col(const float* __restrict__ asum, bf16* __restrict__ patch) {
  int i = blockIdx.x * blockDim.x + threadIdx.x;
  if (i >= NB * NHW * NTP) return;
  int pos = i >> 7, t = i & 127;
  float v = 0.f;
  if (t < 121) {
    int b = pos >> 9, hw = pos & 511, y = hw >> 5, x = hw & 31;
    int dy = t / 11 - 5, dx = t - (t / 11) * 11 - 5;
    int yy = y + dy, xx = x + dx;
    if (yy >= 0 && yy < 16 && xx >= 0 && xx < 32)
      v = asum[b * NHW + yy * 32 + xx];
  }
  patch[i] = (bf16)v;
}

// e[b,hw] = sum_a tanh(q + cov_t + ft) * ew[a] + eb   (one wave per position)
__global__ void k_attn_e(const float* __restrict__ q, const float* __restrict__ covt,
                         const float* __restrict__ ft, const float* __restrict__ ew,
                         const float* __restrict__ eb, float* __restrict__ e) {
  int lane = threadIdx.x & 31;
  int pos  = blockIdx.x * (blockDim.x >> 5) + (threadIdx.x >> 5);
  int b    = pos >> 9;
  const float* cr = covt + pos * NA;
  const float* fr = ft + pos * NA;
  const float* qr = q + b * NA;
  float s = 0.f;
  for (int a = lane; a < NA; a += 32) s += tanhf(qr[a] + cr[a] + fr[a]) * ew[a];
  for (int off = 16; off; off >>= 1) s += __shfl_xor(s, off, 32);
  if (!lane) e[pos] = s + eb[0];
}

// masked softmax over 512 positions, alpha-sum update, alpha output write
__global__ void k_softmax(const float* __restrict__ e, const float* __restrict__ maskS,
                          float* __restrict__ alpha, float* __restrict__ asum,
                          float* __restrict__ was_out, float* __restrict__ alpha_out,
                          int tindex) {
  __shared__ float red[512];
  int b = blockIdx.x, hw = threadIdx.x;
  float v = e[b * NHW + hw];
  red[hw] = v; __syncthreads();
  for (int s = 256; s > 0; s >>= 1) {
    if (hw < s) red[hw] = fmaxf(red[hw], red[hw + s]);
    __syncthreads();
  }
  float m = red[0]; __syncthreads();
  float ee = expf(v - m) * maskS[b * NHW + hw];
  red[hw] = ee; __syncthreads();
  for (int s = 256; s > 0; s >>= 1) {
    if (hw < s) red[hw] += red[hw + s];
    __syncthreads();
  }
  float a = ee / (red[0] + 1e-10f);
  alpha[b * NHW + hw] = a;
  float ns = a + asum[b * NHW + hw];
  asum[b * NHW + hw] = ns;
  if (was_out) was_out[b * NHW + hw] = ns;
  alpha_out[(b * NT + tindex) * NHW + hw] = a;
}

// ctx[b,c] = sum_hw cnn[b,c,hw] * alpha[b,hw]  -> bf16 padded row (B, NCP)
__global__ void k_ctx(const float* __restrict__ cnn, const float* __restrict__ alpha,
                      bf16* __restrict__ ctx_bf) {
  int i = blockIdx.x * blockDim.x + threadIdx.x;
  if (i >= NB * NCP) return;
  int b = i / NCP, c = i - b * NCP;
  float s = 0.f;
  if (c < NC) {
    const float* fp = cnn + (b * NC + c) * NHW;
    const float* ap = alpha + b * NHW;
    for (int hw = 0; hw < NHW; ++hw) s += fp[hw] * ap[hw];
  }
  ctx_bf[i] = (bf16)s;
}

// ------------------------------ host driver --------------------------------

extern "C" void kernel_launch(void* const* d_in, const int* in_sizes, int n_in,
                              void* d_out, int out_size, void* d_ws, size_t ws_size,
                              hipStream_t stream) {
  (void)in_sizes; (void)n_in; (void)out_size; (void)ws_size;
  auto F = [&](int i) { return (const float*)d_in[i]; };

  // inputs: 0 cnn_features, 1 labels, 2 images_mask; params flattened in
  // sorted-key order (jax tree flatten): see index comments below.
  const float* cnn    = F(0);
  const int*   labels = (const int*)d_in[1];
  const float* imask  = F(2);
  // 3 ca.cov_k 4 ca.cov_w 5 ca.e.w 6 ca.e.b 7 ca.feat.w 8 ca.feat.b 9 ca.q.w
  // 10 ca.q.b 11 cconv.w 12 cconv.b 13 ccw.w 14 ccw.b 15 cig.bhh 16 cig.bih
  // 17 cig.whh 18 cig.wih 19 cog.bhh 20 cog.bih 21 cog.whh 22 cog.wih
  // 23 crw.w 24 crw.b 25 csw.w 26 csw.b 27 cww.w 28 cww.b 29 embedding
  // 30 init.w 31 init.b 32 sconv.w 33 sconv.b 34 wa.cov_k 35 wa.cov_w
  // 36 wa.e.w 37 wa.e.b 38 wa.feat.w 39 wa.feat.b 40 wa.q.w 41 wa.q.b
  // 42 wconv.w 43 wconv.b 44 wcw.w 45 wcw.b 46 wew.w 47 wew.b 48 wig.bhh
  // 49 wig.bih 50 wig.whh 51 wig.wih 52 wog.bhh 53 wog.bih 54 wog.whh
  // 55 wog.wih 56 wsw.w 57 wsw.b
  const float* E = F(29);

  // ---- workspace bump allocator ----
  char* wsp = (char*)d_ws;
  auto alloc  = [&](size_t elems, size_t esz) {
    void* r = (void*)wsp;
    wsp += (elems * esz + 255) & ~(size_t)255;
    return r;
  };
  auto AF = [&](size_t n) { return (float*)alloc(n, 4); };
  auto AH = [&](size_t n) { return (bf16*)alloc(n, 2); };

  // bf16 weights (transposed Kp x Np) + data
  bf16* Xc      = AH((size_t)NB * NHW * NCP);
  bf16* ftw_wt  = AH(NCP * NA);   bf16* ftc_wt  = AH(NCP * NA);
  bf16* qw_wt   = AH(NH * NA);    bf16* qc_wt   = AH(NH * NA);
  bf16* fused_w = AH(NTP * NA);   bf16* fused_c = AH(NTP * NA);
  bf16* wig_ih  = AH(NH * NH3);   bf16* wig_hh  = AH(NH * NH3);
  bf16* wog_ih  = AH(NCP * NH3);  bf16* wog_hh  = AH(NH * NH3);
  bf16* cig_ih  = AH(2 * NH * NH3); bf16* cig_hh = AH(NH * NH3);
  bf16* cog_ih  = AH(NCP * NH3);  bf16* cog_hh  = AH(NH * NH3);
  bf16* init_wt = AH(NCP * NH);
  bf16* wsw_t   = AH(NH * 128);   bf16* wew_t = AH(NH * 128);
  bf16* wcw_t   = AH(NCP * 128);  bf16* wconv_t = AH(128 * 128);
  bf16* sconv_t = AH(128 * 32);
  bf16* csw_t   = AH(NH * 128);   bf16* cww_t = AH(NH * 128);
  bf16* crw_t   = AH(NH * 128);   bf16* ccw_t = AH(NCP * 128);
  bf16* cconv_t = AH(128 * 128);
  bf16* avg_bf  = AH(NB * NCP);
  bf16* patch   = AH((size_t)NB * NHW * NTP);
  bf16* phid_bf = AH(NB * NH);    bf16* emb_bf = AH(NB * NH);
  bf16* h1_bf   = AH(NB * NH);    bf16* hid_bf = AH(NB * NH);
  bf16* ctx_bf  = AH(NB * NCP);
  bf16* cat_bf  = AH(NB * 2 * NH); bf16* ce_bf = AH(NB * NH);
  bf16* re_bf   = AH(NB * NH);    bf16* c2ph_bf = AH(NB * NH);
  bf16* c2ph1_bf= AH(NB * NH);    bf16* c2phid_bf = AH(NB * NH);
  bf16* c2pctx_bf = AH(NB * NCP);
  bf16* os_bf   = AH(NB * 128);   bf16* cs_bf = AH(NB * 128);

  // fp32 scratch/state
  float* ft_w   = AF((size_t)NB * NHW * NA);
  float* ft_c   = AF((size_t)NB * NHW * NA);
  float* covt   = AF((size_t)NB * NHW * NA);
  float* ph     = AF((size_t)(NT + 1) * NB * NH);
  float* was    = AF((size_t)(NT + 1) * NB * NHW);
  float* c2ph   = AF(NB * NH);
  float* c2pas  = AF(NB * NHW);
  float* maskS  = AF(NB * NHW);   float* msum = AF(NB);
  float* gi     = AF(NB * NH3);   float* gh   = AF(NB * NH3);
  float* phid   = AF(NB * NH);    float* h1   = AF(NB * NH);
  float* c2ph1  = AF(NB * NH);
  float* qbuf   = AF(NB * NA);    float* ebuf = AF(NB * NHW);
  float* alpha  = AF(NB * NHW);   float* asum_cur = AF(NB * NHW);
  float* os     = AF(NB * 128);   float* cs   = AF(NB * 128);

  // output regions (fp32, concatenated in return order)
  float* outp = (float*)d_out;
  float* wp   = outp;
  float* sp   = wp + (size_t)NB * NT * NV;
  float* wa_o = sp + (size_t)NB * NT * NS;
  float* cp   = wa_o + (size_t)NB * NT * NHW;
  float* ca_o = cp + (size_t)NB * NT * NV;

  auto prep = [&](const float* W, bf16* Wt, int N, int K, int Kp, int Np) {
    int n = Kp * Np;
    k_prep_wt<<<(n + 255) / 256, 256, 0, stream>>>(W, Wt, N, K, Kp, Np);
  };
  auto gemm = [&](const bf16* X, int ldx, const bf16* Wt, int ldw,
                  const float* bias, float* Y, int ldy, int M, int N, int Kp,
                  int mode) {
    int tiles = (M >> 4) * (ldw >> 5);     // 16x32 tiles per wave
    k_gemm<<<(tiles + 7) / 8, 256, 0, stream>>>(X, ldx, Wt, ldw, bias, Y, ldy,
                                                M, N, Kp, mode);
  };
  auto zero = [&](float* p, int n) {
    k_zero<<<256, 256, 0, stream>>>(p, n);
  };

  // ---- state init (deterministic every call) ----
  zero(ph, (NT + 1) * NB * NH);
  zero(was, (NT + 1) * NB * NHW);
  zero(c2ph, NB * NH);
  zero(c2pas, NB * NHW);

  // ---- weight prep ----
  prep(F(38), ftw_wt, NA, NC, NCP, NA);     prep(F(7),  ftc_wt, NA, NC, NCP, NA);
  prep(F(40), qw_wt,  NA, NH, NH,  NA);     prep(F(9),  qc_wt,  NA, NH, NH, NA);
  prep(F(30), init_wt,NH, NC, NCP, NH);
  prep(F(51), wig_ih, NH3, NH, NH, NH3);    prep(F(50), wig_hh, NH3, NH, NH, NH3);
  prep(F(55), wog_ih, NH3, NC, NCP, NH3);   prep(F(54), wog_hh, NH3, NH, NH, NH3);
  prep(F(18), cig_ih, NH3, 2 * NH, 2 * NH, NH3);
  prep(F(17), cig_hh, NH3, NH, NH, NH3);
  prep(F(22), cog_ih, NH3, NC, NCP, NH3);   prep(F(21), cog_hh, NH3, NH, NH, NH3);
  prep(F(56), wsw_t, 128, NH, NH, 128);     prep(F(46), wew_t, 128, NH, NH, 128);
  prep(F(44), wcw_t, 128, NC, NCP, 128);    prep(F(42), wconv_t, NV, 128, 128, 128);
  prep(F(32), sconv_t, NS, 128, 128, 32);
  prep(F(25), csw_t, 128, NH, NH, 128);     prep(F(27), cww_t, 128, NH, NH, 128);
  prep(F(23), crw_t, 128, NH, NH, 128);     prep(F(13), ccw_t, 128, NC, NCP, 128);
  prep(F(11), cconv_t, NV, 128, 128, 128);
  k_fuse_cov<<<(NTP * NA + 255) / 256, 256, 0, stream>>>(F(34), F(35), fused_w);
  k_fuse_cov<<<(NTP * NA + 255) / 256, 256, 0, stream>>>(F(3), F(4), fused_c);

  // ---- feature precompute ----
  {
    int n = NB * NHW * NCP;
    k_xc<<<(n + 255) / 256, 256, 0, stream>>>(cnn, Xc);
  }
  k_mask<<<NB, 512, 0, stream>>>(imask, maskS, msum);
  k_avg<<<(NB * NCP + 255) / 256, 256, 0, stream>>>(cnn, maskS, msum, avg_bf);
  // h0 = tanh(avg @ init.T + b) -> ph[0]
  gemm(avg_bf, NCP, init_wt, NH, F(31), ph, NH, NB, NH, NCP, 2);
  // ft_w / ft_c = feat projections (+bias)
  gemm(Xc, NCP, ftw_wt, NA, F(39), ft_w, NA, NB * NHW, NA, NCP, 0);
  gemm(Xc, NCP, ftc_wt, NA, F(8),  ft_c, NA, NB * NHW, NA, NCP, 0);

  // ---- timestep loop ----
  for (int i = 0; i < NT; ++i) {
    // ===== word (parent->child) branch =====
    k_gather_w<<<NB, 512, 0, stream>>>(labels, ph, was, E, phid, phid_bf,
                                       emb_bf, asum_cur, i);
    gemm(emb_bf, NH, wig_ih, NH3, F(49), gi, NH3, NB, NH3, NH, 0);
    gemm(phid_bf, NH, wig_hh, NH3, F(48), gh, NH3, NB, NH3, NH, 0);
    k_gru<<<16, 256, 0, stream>>>(gi, gh, phid, h1, h1_bf);
    // attention (wa)
    gemm(h1_bf, NH, qw_wt, NA, F(41), qbuf, NA, NB, NA, NH, 0);
    k_im2col<<<(NB * NHW * NTP + 255) / 256, 256, 0, stream>>>(asum_cur, patch);
    gemm(patch, NTP, fused_w, NA, nullptr, covt, NA, NB * NHW, NA, NTP, 0);
    k_attn_e<<<NB * NHW / 8, 256, 0, stream>>>(qbuf, covt, ft_w, F(36), F(37), ebuf);
    k_softmax<<<NB, 512, 0, stream>>>(ebuf, maskS, alpha, asum_cur,
                                      was + (size_t)(i + 1) * NB * NHW, wa_o, i);
    k_ctx<<<(NB * NCP + 255) / 256, 256, 0, stream>>>(cnn, alpha, ctx_bf);
    // hid = GRU(ctx, h1; wog) -> ph[i+1]
    gemm(ctx_bf, NCP, wog_ih, NH3, F(53), gi, NH3, NB, NH3, NCP, 0);
    gemm(h1_bf, NH, wog_hh, NH3, F(52), gh, NH3, NB, NH3, NH, 0);
    k_gru<<<16, 256, 0, stream>>>(gi, gh, h1, ph + (size_t)(i + 1) * NB * NH, hid_bf);
    // out_state = wsw(hid)+wew(emb)+wcw(ctx); probs
    gemm(hid_bf, NH, wsw_t, 128, F(57), os, 128, NB, 128, NH, 0);
    gemm(emb_bf, NH, wew_t, 128, F(47), os, 128, NB, 128, NH, 1);
    gemm(ctx_bf, NCP, wcw_t, 128, F(45), os, 128, NB, 128, NCP, 1);
    k_cvt<<<8, 256, 0, stream>>>(os, os_bf, NB * 128);
    gemm(os_bf, 128, wconv_t, 128, F(43), wp + (size_t)i * NV, NT * NV, NB, NV, 128, 0);
    gemm(os_bf, 128, sconv_t, 32, F(33), sp + (size_t)i * NS, NT * NS, NB, NS, 128, 0);

    // ===== c2p (child->parent) branch =====
    k_gather_c2p<<<NB, 256, 0, stream>>>(labels, E, c2ph, cat_bf, ce_bf, re_bf,
                                         c2ph_bf, i);
    gemm(cat_bf, 2 * NH, cig_ih, NH3, F(16), gi, NH3, NB, NH3, 2 * NH, 0);
    gemm(c2ph_bf, NH, cig_hh, NH3, F(15), gh, NH3, NB, NH3, NH, 0);
    k_gru<<<16, 256, 0, stream>>>(gi, gh, c2ph, c2ph1, c2ph1_bf);
    // c2p_hidden = GRU(ctx, h1; cog)  (uses word-branch ctx/h1, per reference)
    gemm(ctx_bf, NCP, cog_ih, NH3, F(20), gi, NH3, NB, NH3, NCP, 0);
    gemm(h1_bf, NH, cog_hh, NH3, F(19), gh, NH3, NB, NH3, NH, 0);
    k_gru<<<16, 256, 0, stream>>>(gi, gh, h1, c2ph, c2phid_bf);
    // attention (ca)
    gemm(c2ph1_bf, NH, qc_wt, NA, F(10), qbuf, NA, NB, NA, NH, 0);
    k_im2col<<<(NB * NHW * NTP + 255) / 256, 256, 0, stream>>>(c2pas, patch);
    gemm(patch, NTP, fused_c, NA, nullptr, covt, NA, NB * NHW, NA, NTP, 0);
    k_attn_e<<<NB * NHW / 8, 256, 0, stream>>>(qbuf, covt, ft_c, F(5), F(6), ebuf);
    k_softmax<<<NB, 512, 0, stream>>>(ebuf, maskS, alpha, c2pas, nullptr, ca_o,
                                      NT - 1 - i);
    k_ctx<<<(NB * NCP + 255) / 256, 256, 0, stream>>>(cnn, alpha, c2pctx_bf);
    // c2p_state = csw(c2p_hidden)+cww(ce)+crw(re)+ccw(c2p_ctx); probs
    gemm(c2phid_bf, NH, csw_t, 128, F(26), cs, 128, NB, 128, NH, 0);
    gemm(ce_bf, NH, cww_t, 128, F(28), cs, 128, NB, 128, NH, 1);
    gemm(re_bf, NH, crw_t, 128, F(24), cs, 128, NB, 128, NH, 1);
    gemm(c2pctx_bf, NCP, ccw_t, 128, F(14), cs, 128, NB, 128, NCP, 1);
    k_cvt<<<8, 256, 0, stream>>>(cs, cs_bf, NB * 128);
    gemm(cs_bf, 128, cconv_t, 128, F(12), cp + (size_t)(NT - 1 - i) * NV, NT * NV,
         NB, NV, 128, 0);
  }
}
